// MultiheadAttention_56985626083997
// MI455X (gfx1250) — matrix operand order repaired and checked by reading, not compiled, not run on hardware
//
#include <hip/hip_runtime.h>
#include <hip/hip_bf16.h>
#include <cstddef>

// ---------------------------------------------------------------------------
// MI455X (gfx1250) fused MHA: bf16 WMMA GEMMs + flash attention,
// with async global->LDS staging where the copy needs no conversion.
// Shapes fixed by the reference: B=2, T=2048, D=1024, H=16, P=64, DK=64.
// ---------------------------------------------------------------------------

typedef __bf16 v16bf __attribute__((ext_vector_type(16)));
typedef __bf16 v8bf  __attribute__((ext_vector_type(8)));
typedef float  v8f   __attribute__((ext_vector_type(8)));
typedef int    v4i   __attribute__((ext_vector_type(4)));

constexpr int Bc  = 2;
constexpr int Tc  = 2048;
constexpr int Dc  = 1024;
constexpr int Hc  = 16;
constexpr int Pc  = 64;
constexpr int DKc = 64;

#if defined(__AMDGCN__) && __has_builtin(__builtin_amdgcn_global_load_async_to_lds_b128)
#define HAVE_ASYNC_LDS 1
#else
#define HAVE_ASYNC_LDS 0
#endif

// ---- helpers --------------------------------------------------------------

// Native conversion: lowers to v_cvt_pk_bf16_f32 pairs on gfx1250.
__device__ __forceinline__ __bf16 f2bf(float f) { return (__bf16)f; }

// 16-byte global -> LDS copy; async DMA (ASYNCcnt) when available.
__device__ __forceinline__ void copy16_g2l(const __bf16* gsrc, __bf16* ldst) {
#if HAVE_ASYNC_LDS
  __builtin_amdgcn_global_load_async_to_lds_b128(
      (__attribute__((address_space(1))) v4i*)gsrc,
      (__attribute__((address_space(3))) v4i*)ldst, 0, 0);
#else
  *(v8bf*)ldst = *(const v8bf*)gsrc;
#endif
}

__device__ __forceinline__ void async_wait() {
#if HAVE_ASYNC_LDS
#if __has_builtin(__builtin_amdgcn_s_wait_asynccnt)
  __builtin_amdgcn_s_wait_asynccnt(0);
#else
  asm volatile("s_wait_asynccnt 0" ::: "memory");
#endif
#endif
}

// Build a 16-element bf16 fragment from two contiguous 8-element (16-byte) chunks.
__device__ __forceinline__ v16bf ld_frag(const __bf16* p0, const __bf16* p1) {
  v8bf a = *(const v8bf*)p0;
  v8bf c = *(const v8bf*)p1;
  v16bf r;
#pragma unroll
  for (int i = 0; i < 8; ++i) { r[i] = a[i]; r[i + 8] = c[i]; }
  return r;
}

__device__ __forceinline__ v8f zero8() {
  v8f z;
#pragma unroll
  for (int i = 0; i < 8; ++i) z[i] = 0.0f;
  return z;
}

__device__ __forceinline__ v8f wmma_bf16(v16bf a, v16bf b, v8f c) {
  // v_wmma_f32_16x16x32_bf16 : D = A(16x32) * B(32x16) + C(16x16 f32)
  return __builtin_amdgcn_wmma_f32_16x16x32_bf16(
      /*neg_a=*/false, a, /*neg_b=*/false, b,
      /*c_mod=*/(short)0, c, /*reuse_a=*/false, /*reuse_b=*/false);
}

// ---- kernel 0: fp32 prefix K/V -> bf16 head-major K/V rows [0, P) ----------

__global__ __launch_bounds__(256) void prefix_copy_kernel(
    const float* __restrict__ pk, const float* __restrict__ pv,
    __bf16* __restrict__ Kw, __bf16* __restrict__ Vw) {
  int idx = blockIdx.x * blockDim.x + threadIdx.x;  // 0 .. 2*B*H*P*DK-1
  const int n = Bc * Hc * Pc * DKc;
  const float* src = (idx < n) ? pk : pv;
  __bf16* dst = (idx < n) ? Kw : Vw;
  int i = (idx < n) ? idx : idx - n;
  int dk = i % DKc;
  int p  = (i / DKc) % Pc;
  int bh = i / (DKc * Pc);
  dst[((size_t)bh * (Pc + Tc) + p) * DKc + dk] = f2bf(src[i]);
}

// ---- kernel 1: generic bf16 WMMA GEMM:  Out = A(MxK) * W^T(KxN) + bias -----
// OUT_MODE 0: bf16 store, head-major remap ((b,h,row_off+t,dk)), value*(alpha)
// OUT_MODE 1: f32 store row-major to dst_f (final projection)

template <int A_IS_F32, int OUT_MODE>
__global__ __launch_bounds__(256) void gemm_wmma_kernel(
    const void* __restrict__ Aptr,
    const float* __restrict__ W, const float* __restrict__ bias,
    int M, int N, int K, float alpha,
    __bf16* __restrict__ dst_bf, float* __restrict__ dst_f,
    int rowstride, int rowoff) {
  __shared__ __align__(16) __bf16 As[128][32];  // [m][k]
  __shared__ __align__(16) __bf16 Bs[64][32];   // [n][k]  (== rows of W)

  const int tid = threadIdx.x;
  const int m0 = blockIdx.x * 128;
  const int n0 = blockIdx.y * 64;
  const int w = tid >> 5;          // wave 0..7 -> owns M rows [w*16, w*16+16)
  const int lane = tid & 31;
  const int r = lane & 15;
  const int hi = lane >> 4;

  const int arow = tid >> 1;
  const int acol = (tid & 1) * 16;
  const int bn = tid >> 2;
  const int bk = (tid & 3) * 8;

  v8f acc[4];
#pragma unroll
  for (int nt = 0; nt < 4; ++nt) acc[nt] = zero8();

  for (int k0 = 0; k0 < K; k0 += 32) {
    // stage A tile (128x32): each thread covers 16 contiguous elements
    if constexpr (A_IS_F32) {
      const float* A = (const float*)Aptr + (size_t)(m0 + arow) * K + k0 + acol;
      if (k0 + 32 < K) __builtin_prefetch(A + 32, 0, 3);  // global_prefetch_b8
#pragma unroll
      for (int c = 0; c < 4; ++c) {
        const float4 t = *(const float4*)(A + c * 4);
        As[arow][acol + c * 4 + 0] = f2bf(t.x);
        As[arow][acol + c * 4 + 1] = f2bf(t.y);
        As[arow][acol + c * 4 + 2] = f2bf(t.z);
        As[arow][acol + c * 4 + 3] = f2bf(t.w);
      }
    } else {
      // pure copy: async LDS-DMA path (global_load_async_to_lds_b128)
      const __bf16* A = (const __bf16*)Aptr + (size_t)(m0 + arow) * K + k0 + acol;
      if (k0 + 32 < K) __builtin_prefetch(A + 32, 0, 3);
      copy16_g2l(A, &As[arow][acol]);
      copy16_g2l(A + 8, &As[arow][acol + 8]);
    }
    // stage B tile (64 n-rows x 32 k): W is row-major [n][k] already
    {
      const float* Wp = W + (size_t)(n0 + bn) * K + k0 + bk;
      if (k0 + 32 < K) __builtin_prefetch(Wp + 32, 0, 3);
      const float4 t0 = *(const float4*)(Wp);
      const float4 t1 = *(const float4*)(Wp + 4);
      Bs[bn][bk + 0] = f2bf(t0.x);
      Bs[bn][bk + 1] = f2bf(t0.y);
      Bs[bn][bk + 2] = f2bf(t0.z);
      Bs[bn][bk + 3] = f2bf(t0.w);
      Bs[bn][bk + 4] = f2bf(t1.x);
      Bs[bn][bk + 5] = f2bf(t1.y);
      Bs[bn][bk + 6] = f2bf(t1.z);
      Bs[bn][bk + 7] = f2bf(t1.w);
    }
    if constexpr (!A_IS_F32) async_wait();
    __syncthreads();

    // A fragment (16x32, lane r = row M): K chunks hi*8.. and 16+hi*8..
    v16bf af = ld_frag(&As[w * 16 + r][hi * 8], &As[w * 16 + r][16 + hi * 8]);
#pragma unroll
    for (int nt = 0; nt < 4; ++nt) {
      // B fragment (32x16, lane r = col N): K chunks hi*8.. and 16+hi*8..
      v16bf bf = ld_frag(&Bs[nt * 16 + r][hi * 8], &Bs[nt * 16 + r][16 + hi * 8]);
      acc[nt] = wmma_bf16(af, bf, acc[nt]);
    }
    __syncthreads();
  }

  // epilogue: C/D layout -> slot i is (M = i + 8*hi, N = r)
#pragma unroll
  for (int nt = 0; nt < 4; ++nt) {
    const int n = n0 + nt * 16 + r;
    const float bv = bias[n];
#pragma unroll
    for (int i = 0; i < 8; ++i) {
      const int m = m0 + w * 16 + i + hi * 8;
      const float val = (acc[nt][i] + bv) * alpha;
      if constexpr (OUT_MODE == 0) {
        const int b = m / Tc, t = m % Tc;
        const int h = n / DKc, dk = n % DKc;
        dst_bf[((size_t)(b * Hc + h) * rowstride + rowoff + t) * DKc + dk] = f2bf(val);
      } else {
        dst_f[(size_t)m * N + n] = val;
      }
    }
  }
}

// ---- kernel 2: flash attention per (b,h), 64 q-rows per block --------------
// Qw: [B*H][T][DK] bf16 (pre-scaled by 1/sqrt(DK))
// Kw/Vw: [B*H][P+T][DK] bf16 (prefix in rows [0,P))
// Ow: [B*T][D] bf16 token-major (input to the final projection)

__global__ __launch_bounds__(128) void flash_attn_kernel(
    const __bf16* __restrict__ Qw, const __bf16* __restrict__ Kw,
    const __bf16* __restrict__ Vw, const unsigned char* __restrict__ kpm,
    __bf16* __restrict__ Ow) {
  __shared__ __align__(16) __bf16 Ks[32][64];      // [kv_local][dk]
  __shared__ __align__(16) __bf16 Vt[64][32];      // [dk][kv_local] (transposed)
  __shared__ __align__(16) __bf16 Ps[4][16][32];   // per-wave probs [m][kv_local]

  const int q0 = blockIdx.x * 64;
  const int bh = blockIdx.y;
  const int b = bh / Hc;
  const int h = bh % Hc;
  const int tid = threadIdx.x;
  const int w = tid >> 5;
  const int lane = tid & 31;
  const int r = lane & 15;
  const int hi = lane >> 4;

  // Q fragments for this wave's 16 rows (lane r -> row), DK=64 -> 2 fragments
  const __bf16* qrow = Qw + ((size_t)bh * Tc + q0 + w * 16 + r) * DKc;
  const v16bf qf0 = ld_frag(qrow + hi * 8, qrow + 16 + hi * 8);
  const v16bf qf1 = ld_frag(qrow + 32 + hi * 8, qrow + 48 + hi * 8);

  int tq[8];
  float qvmask[8];  // 1 if query token valid, else 0 (used as bool)
#pragma unroll
  for (int i = 0; i < 8; ++i) {
    tq[i] = q0 + w * 16 + i + hi * 8;
    qvmask[i] = (float)kpm[b * Tc + tq[i]];
  }

  float mrow[8], lrow[8];
  v8f acc[4];
#pragma unroll
  for (int i = 0; i < 8; ++i) { mrow[i] = -3.0e38f; lrow[i] = 0.0f; }
#pragma unroll
  for (int nt = 0; nt < 4; ++nt) acc[nt] = zero8();

  const size_t krowbase = (size_t)bh * (Pc + Tc) * DKc;
  const int nkv = Pc + q0 + 64;  // causal: keys up to last row of this q-tile

  const int kvl = tid >> 2;
  const int c = (tid & 3) * 16;

  for (int kv0 = 0; kv0 < nkv; kv0 += 32) {
    // --- stage K (async DMA, row-major) and V (transposed, VALU path) ---
    {
      const __bf16* ks = Kw + krowbase + (size_t)(kv0 + kvl) * DKc + c;
      copy16_g2l(ks, &Ks[kvl][c]);
      copy16_g2l(ks + 8, &Ks[kvl][c + 8]);
      const __bf16* vs = Vw + krowbase + (size_t)(kv0 + kvl) * DKc + c;
#pragma unroll
      for (int j = 0; j < 16; ++j) Vt[c + j][kvl] = vs[j];
      if (kv0 + 32 < nkv) {  // prefetch next KV tile (global_prefetch_b8)
        __builtin_prefetch(ks + 32 * DKc, 0, 3);
        __builtin_prefetch(vs + 32 * DKc, 0, 3);
      }
    }
    async_wait();
    __syncthreads();

    // --- logits: Q(16x64) * K^T(64x32) as 2 col-tiles x 2 K-chunks ---
    float pl[2][8];
    float mcur[8];
#pragma unroll
    for (int i = 0; i < 8; ++i) mcur[i] = -3.0e38f;
#pragma unroll
    for (int nt = 0; nt < 2; ++nt) {
      v16bf b0 = ld_frag(&Ks[nt * 16 + r][hi * 8], &Ks[nt * 16 + r][16 + hi * 8]);
      v16bf b1 = ld_frag(&Ks[nt * 16 + r][32 + hi * 8], &Ks[nt * 16 + r][48 + hi * 8]);
      v8f z = zero8();
      z = wmma_bf16(qf0, b0, z);
      z = wmma_bf16(qf1, b1, z);
      // mask (column tk is the same for all 8 row-slots of this lane)
      const int tk = kv0 + nt * 16 + r;
      const int tkr = tk - Pc;
      const bool kallow = (tk < Pc) || (kpm[b * Tc + tkr] != 0);
#pragma unroll
      for (int i = 0; i < 8; ++i) {
        float lv = z[i];
        const bool allow =
            (qvmask[i] != 0.0f) && ((tk < Pc) || (kallow && (tkr <= tq[i])));
        lv = allow ? lv : -1.0e9f;
        pl[nt][i] = lv;
        mcur[i] = fmaxf(mcur[i], lv);
      }
    }
    // row-max across the 16 lanes holding each row (xor masks stay in-group)
#pragma unroll
    for (int msk = 1; msk <= 8; msk <<= 1)
#pragma unroll
      for (int i = 0; i < 8; ++i)
        mcur[i] = fmaxf(mcur[i], __shfl_xor(mcur[i], msk, 32));

    float scale[8], psum[8];
#pragma unroll
    for (int i = 0; i < 8; ++i) {
      const float mn = fmaxf(mrow[i], mcur[i]);
      scale[i] = __expf(mrow[i] - mn);
      mrow[i] = mn;
      const float p0 = __expf(pl[0][i] - mn);
      const float p1 = __expf(pl[1][i] - mn);
      pl[0][i] = p0;
      pl[1][i] = p1;
      psum[i] = p0 + p1;
    }
#pragma unroll
    for (int msk = 1; msk <= 8; msk <<= 1)
#pragma unroll
      for (int i = 0; i < 8; ++i) psum[i] += __shfl_xor(psum[i], msk, 32);
#pragma unroll
    for (int i = 0; i < 8; ++i) lrow[i] = lrow[i] * scale[i] + psum[i];
#pragma unroll
    for (int nt = 0; nt < 4; ++nt)
#pragma unroll
      for (int i = 0; i < 8; ++i) acc[nt][i] *= scale[i];

    // C-layout probs -> A-layout via per-wave LDS tile
#pragma unroll
    for (int nt = 0; nt < 2; ++nt)
#pragma unroll
      for (int i = 0; i < 8; ++i)
        Ps[w][i + hi * 8][nt * 16 + r] = f2bf(pl[nt][i]);
    __syncthreads();

    // --- out += P(16x32) * V(32x64) : 4 dk col-tiles ---
    v16bf pf = ld_frag(&Ps[w][r][hi * 8], &Ps[w][r][16 + hi * 8]);
#pragma unroll
    for (int nt = 0; nt < 4; ++nt) {
      v16bf vf = ld_frag(&Vt[nt * 16 + r][hi * 8], &Vt[nt * 16 + r][16 + hi * 8]);
      acc[nt] = wmma_bf16(pf, vf, acc[nt]);
    }
    __syncthreads();
  }

  // epilogue: normalize and store token-major bf16
#pragma unroll
  for (int nt = 0; nt < 4; ++nt) {
    const int dk = nt * 16 + r;
#pragma unroll
    for (int i = 0; i < 8; ++i) {
      const float val = acc[nt][i] / lrow[i];
      Ow[((size_t)b * Tc + tq[i]) * Dc + h * DKc + dk] = f2bf(val);
    }
  }
}

// ---------------------------------------------------------------------------

extern "C" void kernel_launch(void* const* d_in, const int* in_sizes, int n_in,
                              void* d_out, int out_size, void* d_ws, size_t ws_size,
                              hipStream_t stream) {
  (void)in_sizes; (void)n_in; (void)out_size; (void)ws_size;
  const float* q  = (const float*)d_in[0];
  const float* k  = (const float*)d_in[1];
  const float* v  = (const float*)d_in[2];
  const float* pk = (const float*)d_in[3];
  const float* pv = (const float*)d_in[4];
  const unsigned char* kpm = (const unsigned char*)d_in[5];  // jnp bool = 1 byte
  const float* Wq = (const float*)d_in[6];
  const float* bq = (const float*)d_in[7];
  const float* Wk = (const float*)d_in[8];
  const float* bk = (const float*)d_in[9];
  const float* Wv = (const float*)d_in[10];
  const float* bv = (const float*)d_in[11];
  const float* Wo = (const float*)d_in[12];
  const float* bo = (const float*)d_in[13];
  float* out = (float*)d_out;

  // workspace carve-up (bf16): Q | K | V | attn-out  (~33 MB total)
  __bf16* Qw = (__bf16*)d_ws;
  __bf16* Kw = Qw + (size_t)Bc * Hc * Tc * DKc;
  __bf16* Vw = Kw + (size_t)Bc * Hc * (Pc + Tc) * DKc;
  __bf16* AO = Vw + (size_t)Bc * Hc * (Pc + Tc) * DKc;

  const int Mrows = Bc * Tc;  // 4096
  const dim3 ggrid(Mrows / 128, Dc / 64);
  const float inv_sqrt_dk = 0.125f;  // 1/sqrt(64)

  // prefix K/V fp32 -> bf16 into rows [0, P)
  {
    const int n = 2 * Bc * Hc * Pc * DKc;
    prefix_copy_kernel<<<n / 256, 256, 0, stream>>>(pk, pv, Kw, Vw);
  }
  // projections (scale folded into Q)
  gemm_wmma_kernel<1, 0><<<ggrid, 256, 0, stream>>>(q, Wq, bq, Mrows, Dc, Dc,
                                                    inv_sqrt_dk, Qw, nullptr, Tc, 0);
  gemm_wmma_kernel<1, 0><<<ggrid, 256, 0, stream>>>(k, Wk, bk, Mrows, Dc, Dc,
                                                    1.0f, Kw, nullptr, Pc + Tc, Pc);
  gemm_wmma_kernel<1, 0><<<ggrid, 256, 0, stream>>>(v, Wv, bv, Mrows, Dc, Dc,
                                                    1.0f, Vw, nullptr, Pc + Tc, Pc);
  // attention
  flash_attn_kernel<<<dim3(Tc / 64, Bc * Hc), 128, 0, stream>>>(Qw, Kw, Vw, kpm, AO);
  // output projection (f32 + bias)
  gemm_wmma_kernel<0, 1><<<ggrid, 256, 0, stream>>>(AO, Wo, bo, Mrows, Dc, Dc,
                                                    1.0f, nullptr, out, 0, 0);
}
